// SSDP_v2_65558380806639
// MI455X (gfx1250) — compile-verified, matrix operand order checked by reading, not run on hardware
//
#include <hip/hip_runtime.h>
#include <hip/hip_bf16.h>

typedef __attribute__((ext_vector_type(16))) _Float16 v16h;
typedef __attribute__((ext_vector_type(8)))  float    v8f;

#define NPTS 512
#define LRELU(x) ((x) >= 0.f ? (x) : 0.2f * (x))

__device__ __forceinline__ int kb_map(int v) {
  return (v < 4) ? (2 * v) : (16 + 2 * (v - 4));
}

// ---------------------------------------------------------------------------
// K1: P/Q rank factors of the edge conv.  P = Wd @ X, Q = (Ws - Wd) @ X.
// ---------------------------------------------------------------------------
__global__ __launch_bounds__(256) void pq_kernel(
    const float* __restrict__ X,     // 64 x 512 (part_feature)
    const float* __restrict__ We1,   // 64 x 128
    const float* __restrict__ We1s,  // 64 x 128
    float* __restrict__ P1, float* __restrict__ Q1,
    float* __restrict__ P2, float* __restrict__ Q2) {
  int id  = blockIdx.x * 256 + threadIdx.x;  // 0 .. 131071
  int mat = id >> 15;
  int rem = id & 32767;
  int o = rem >> 9;
  int n = rem & 511;
  const float* W = (mat < 2) ? We1 : We1s;
  float acc = 0.f;
  for (int c = 0; c < 64; ++c) {
    float w = ((mat & 1) == 0) ? W[o * 128 + c]
                               : (W[o * 128 + 64 + c] - W[o * 128 + c]);
    acc += w * X[c * 512 + n];
  }
  float* dst = (mat == 0) ? P1 : (mat == 1) ? Q1 : (mat == 2) ? P2 : Q2;
  dst[o * 512 + n] = acc;
}

// ---------------------------------------------------------------------------
// K2: per-row e1 = lrelu(bn(P[o,j]+Q[o,i])), per-head softmax attention,
//     store attention-weighted e1 as f16 in [i][j][o] layout for WMMA B-feeds.
// ---------------------------------------------------------------------------
__global__ __launch_bounds__(256) void e1_attn_kernel(
    const float* __restrict__ P1, const float* __restrict__ Q1,
    const float* __restrict__ P2, const float* __restrict__ Q2,
    const float* __restrict__ bn1, const float* __restrict__ bn2,
    const float* __restrict__ wa1, const float* __restrict__ wa2,
    _Float16* __restrict__ out1, _Float16* __restrict__ out2) {
  int i = blockIdx.x, br = blockIdx.y;
  const float* P   = br ? P2  : P1;
  const float* Q   = br ? Q2  : Q1;
  const float* bn  = br ? bn2 : bn1;
  const float* wav = br ? wa2 : wa1;
  _Float16* out    = br ? out2 : out1;

  __shared__ float sS[64], sT[64], sQ[64], wa[16];
  __shared__ float lg[4 * NPTS];
  __shared__ float red[256];
  __shared__ float hstat[8];  // [0..3]=max, [4..7]=1/sum
  int tid = threadIdx.x;

  if (tid < 64) {
    float g = bn[tid], b = bn[64 + tid], m = bn[128 + tid], v = bn[192 + tid];
    float s = g * rsqrtf(v + 1e-5f);
    sS[tid] = s; sT[tid] = b - m * s;
    sQ[tid] = Q[tid * 512 + i];
  }
  if (tid < 16) wa[tid] = wav[tid];
  __syncthreads();

  // Phase A: attention logits per head
  for (int jj = tid; jj < NPTS; jj += 256) {
    float acc[4] = {0.f, 0.f, 0.f, 0.f};
#pragma unroll
    for (int o = 0; o < 64; ++o) {
      float x = sS[o] * (P[o * 512 + jj] + sQ[o]) + sT[o];
      x = LRELU(x);            // e1 value
      float y = LRELU(x);      // lrelu applied again inside _attn
      acc[o & 3] += wa[o >> 2] * y;
    }
    lg[0 * NPTS + jj] = acc[0];
    lg[1 * NPTS + jj] = acc[1];
    lg[2 * NPTS + jj] = acc[2];
    lg[3 * NPTS + jj] = acc[3];
  }
  __syncthreads();

  // Phase B: softmax stats (max, 1/sum) per head
  for (int h = 0; h < 4; ++h) {
    float a = lg[h * NPTS + tid], b = lg[h * NPTS + tid + 256];
    red[tid] = fmaxf(a, b);
    __syncthreads();
    for (int s = 128; s > 0; s >>= 1) {
      if (tid < s) red[tid] = fmaxf(red[tid], red[tid + s]);
      __syncthreads();
    }
    if (tid == 0) hstat[h] = red[0];
    __syncthreads();
    float mx = hstat[h];
    red[tid] = __expf(a - mx) + __expf(b - mx);
    __syncthreads();
    for (int s = 128; s > 0; s >>= 1) {
      if (tid < s) red[tid] += red[tid + s];
      __syncthreads();
    }
    if (tid == 0) hstat[4 + h] = 1.f / red[0];
    __syncthreads();
  }

  // Phase C: recompute e1, weight by attention, store f16 (contiguous 128B/j)
  for (int jj = tid; jj < NPTS; jj += 256) {
    float at[4];
#pragma unroll
    for (int h = 0; h < 4; ++h)
      at[h] = __expf(lg[h * NPTS + jj] - hstat[h]) * hstat[4 + h];
    __align__(16) _Float16 tmp[64];
#pragma unroll
    for (int o = 0; o < 64; ++o) {
      float x = sS[o] * (P[o * 512 + jj] + sQ[o]) + sT[o];
      x = LRELU(x);
      tmp[o] = (_Float16)(at[o & 3] * x);
    }
    uint4* dst = (uint4*)(out + ((size_t)i * 512 + jj) * 64);
    const uint4* src = (const uint4*)tmp;
#pragma unroll
    for (int q = 0; q < 8; ++q) dst[q] = src[q];
  }
}

// ---------------------------------------------------------------------------
// K3: hop branch, fused per row: WMMA (32x64)@(64x512) -> bn/lrelu -> LDS f16
//     -> WMMA (10x32)@(32x512) -> bn/lrelu -> hop_logits row.
// ---------------------------------------------------------------------------
__global__ __launch_bounds__(256) void hop_kernel(
    const _Float16* __restrict__ e1w1, const _Float16* __restrict__ e1w2,
    const float* __restrict__ Wa1, const float* __restrict__ bna1,
    const float* __restrict__ Wb1, const float* __restrict__ bnb1,
    const float* __restrict__ Wa2, const float* __restrict__ bna2,
    const float* __restrict__ Wb2, const float* __restrict__ bnb2,
    float* __restrict__ outbase) {
  int i = blockIdx.x, br = blockIdx.y;
  const _Float16* e1w = br ? e1w2 : e1w1;
  const float* Wa  = br ? Wa2  : Wa1;
  const float* bna = br ? bna2 : bna1;
  const float* Wb  = br ? Wb2  : Wb1;
  const float* bnb = br ? bnb2 : bnb1;
  float* outp = outbase + (size_t)br * (10u * 512u * 512u);

  __shared__ _Float16 slab[NPTS * 32];  // 32 KB: lrelu(bn(Wa@e1w)) as f16
  __shared__ float sA[32], tA[32], sB[10], tB[10];
  int tid = threadIdx.x;
  if (tid < 32) {
    float g = bna[tid], b = bna[32 + tid], m = bna[64 + tid], v = bna[96 + tid];
    float s = g * rsqrtf(v + 1e-5f);
    sA[tid] = s; tA[tid] = b - m * s;
  }
  if (tid < 10) {
    float g = bnb[tid], b = bnb[10 + tid], m = bnb[20 + tid], v = bnb[30 + tid];
    float s = g * rsqrtf(v + 1e-5f);
    sB[tid] = s; tB[tid] = b - m * s;
  }
  __syncthreads();

  int lane = tid & 31, w = tid >> 5;
  int mrow = lane & 15, kh = lane >> 4;
  const _Float16* rowslab = e1w + (size_t)i * (512 * 64);

  // GEMM1: 2 mtiles x 32 ntiles over 8 waves (8 tiles each), K=64 (2 steps)
  for (int t8 = 0; t8 < 8; ++t8) {
    int tile = w * 8 + t8;
    int mt = tile >> 5, nt = tile & 31;
    const _Float16* col = rowslab + (size_t)(nt * 16 + mrow) * 64;
    v8f c = {};
#pragma unroll
    for (int kit = 0; kit < 2; ++kit) {
      v16h a, b;
      int r = mt * 16 + mrow;
#pragma unroll
      for (int v2 = 0; v2 < 8; ++v2) {
        int k = kit * 32 + kb_map(v2) + 8 * kh;
        a[2 * v2]     = (_Float16)Wa[r * 64 + k];
        a[2 * v2 + 1] = (_Float16)Wa[r * 64 + k + 1];
        b[2 * v2]     = col[k];
        b[2 * v2 + 1] = col[k + 1];
      }
      c = __builtin_amdgcn_wmma_f32_16x16x32_f16(false, a, false, b,
                                                 (short)0, c, false, false);
    }
    int j = nt * 16 + mrow;
    int cb = mt * 16 + 8 * kh;
#pragma unroll
    for (int v2 = 0; v2 < 8; ++v2) {
      int ch = cb + v2;
      float x = c[v2] * sA[ch] + tA[ch];
      x = LRELU(x);
      slab[j * 32 + ch] = (_Float16)x;
    }
  }
  __syncthreads();

  // GEMM2: M=10 (padded to 16), K=32, 32 ntiles over 8 waves (4 each)
  for (int q = 0; q < 4; ++q) {
    int nt = w * 4 + q;
    v16h a, b;
    const _Float16* col = slab + (nt * 16 + mrow) * 32;
#pragma unroll
    for (int v2 = 0; v2 < 8; ++v2) {
      int k = kb_map(v2) + 8 * kh;
      float w0 = (mrow < 10) ? Wb[mrow * 32 + k]     : 0.f;
      float w1 = (mrow < 10) ? Wb[mrow * 32 + k + 1] : 0.f;
      a[2 * v2]     = (_Float16)w0;
      a[2 * v2 + 1] = (_Float16)w1;
      b[2 * v2]     = col[k];
      b[2 * v2 + 1] = col[k + 1];
    }
    v8f c = {};
    c = __builtin_amdgcn_wmma_f32_16x16x32_f16(false, a, false, b,
                                               (short)0, c, false, false);
    int j = nt * 16 + mrow;
#pragma unroll
    for (int v2 = 0; v2 < 8; ++v2) {
      int ch = 8 * kh + v2;
      if (ch < 10) {
        float x = c[v2] * sB[ch] + tB[ch];
        x = LRELU(x);
        outp[((size_t)ch * 512 + i) * 512 + j] = x;
      }
    }
  }
}

// ---------------------------------------------------------------------------
// K4: e2 branch, fused per (row, head): WMMA (16x64)@(64x512) -> bn/lrelu
//     -> gaussian-gated softmax over j -> weighted mean -> g[o,i].
// ---------------------------------------------------------------------------
__global__ __launch_bounds__(256) void e2_kernel(
    const _Float16* __restrict__ e1w1, const _Float16* __restrict__ e1w2,
    const float* __restrict__ We2,  const float* __restrict__ bne2,
    const float* __restrict__ wa2,
    const float* __restrict__ We2s, const float* __restrict__ bne2s,
    const float* __restrict__ wa2s,
    const int* __restrict__ dist1, const int* __restrict__ dist2,
    float* __restrict__ gout) {
  int i = blockIdx.x, h = blockIdx.y;
  __shared__ float e2s[16 * NPTS];  // 32 KB
  __shared__ float lg[NPTS];
  __shared__ float red[256];
  __shared__ float s_[16], t_[16], wa[16];
  __shared__ float gacc[16];
  __shared__ float stat[2];
  int tid = threadIdx.x, lane = tid & 31, w = tid >> 5;
  int mrow = lane & 15, kh = lane >> 4;
  if (tid < 16) gacc[tid] = 0.f;

  for (int br = 0; br < 2; ++br) {
    const _Float16* e1w = br ? e1w2 : e1w1;
    const float* W   = br ? We2s  : We2;
    const float* bn  = br ? bne2s : bne2;
    const float* wav = br ? wa2s  : wa2;
    const int* dist  = br ? dist2 : dist1;

    if (tid < 16) {
      int ch = tid * 4 + h;  // o = d*HEADS + h
      float g = bn[ch], b = bn[64 + ch], m = bn[128 + ch], v = bn[192 + ch];
      float s = g * rsqrtf(v + 1e-5f);
      s_[tid] = s; t_[tid] = b - m * s;
      wa[tid] = wav[tid];
    }
    __syncthreads();

    const _Float16* rowslab = e1w + (size_t)i * (512 * 64);
    // GEMM: M=16 (d, gathered rows o=d*4+h), K=64 (2 steps), 32 ntiles / 8 waves
    for (int q = 0; q < 4; ++q) {
      int nt = w * 4 + q;
      const _Float16* col = rowslab + (size_t)(nt * 16 + mrow) * 64;
      v8f c = {};
#pragma unroll
      for (int kit = 0; kit < 2; ++kit) {
        v16h a, b;
        int r = mrow * 4 + h;  // gathered weight row
#pragma unroll
        for (int v2 = 0; v2 < 8; ++v2) {
          int k = kit * 32 + kb_map(v2) + 8 * kh;
          a[2 * v2]     = (_Float16)W[r * 64 + k];
          a[2 * v2 + 1] = (_Float16)W[r * 64 + k + 1];
          b[2 * v2]     = col[k];
          b[2 * v2 + 1] = col[k + 1];
        }
        c = __builtin_amdgcn_wmma_f32_16x16x32_f16(false, a, false, b,
                                                   (short)0, c, false, false);
      }
      int j = nt * 16 + mrow;
#pragma unroll
      for (int v2 = 0; v2 < 8; ++v2) {
        int d = 8 * kh + v2;
        float x = c[v2] * s_[d] + t_[d];
        e2s[d * NPTS + j] = LRELU(x);
      }
    }
    __syncthreads();

    // attention logits with gaussian gate
    for (int jj = tid; jj < NPTS; jj += 256) {
      float xd = (float)dist[i * 512 + jj];
      float gt = __expf(-0.5f * xd * xd) * 0.3989422804014327f;
      float acc = 0.f;
#pragma unroll
      for (int d = 0; d < 16; ++d) {
        float e = gt * e2s[d * NPTS + jj];
        acc += wa[d] * LRELU(e);
      }
      lg[jj] = acc;
    }
    __syncthreads();

    float a0 = lg[tid], a1 = lg[tid + 256];
    red[tid] = fmaxf(a0, a1);
    __syncthreads();
    for (int s = 128; s > 0; s >>= 1) {
      if (tid < s) red[tid] = fmaxf(red[tid], red[tid + s]);
      __syncthreads();
    }
    if (tid == 0) stat[0] = red[0];
    __syncthreads();
    float mx = stat[0];
    red[tid] = __expf(a0 - mx) + __expf(a1 - mx);
    __syncthreads();
    for (int s = 128; s > 0; s >>= 1) {
      if (tid < s) red[tid] += red[tid + s];
      __syncthreads();
    }
    if (tid == 0) stat[1] = 1.f / red[0];
    __syncthreads();
    float inv = stat[1];
    lg[tid]       = __expf(a0 - mx) * inv;
    lg[tid + 256] = __expf(a1 - mx) * inv;
    __syncthreads();

    // weighted mean over j per d, accumulate into gacc
    for (int d = 0; d < 16; ++d) {
      float part = lg[tid] * e2s[d * NPTS + tid] +
                   lg[tid + 256] * e2s[d * NPTS + tid + 256];
      red[tid] = part;
      __syncthreads();
      for (int s = 128; s > 0; s >>= 1) {
        if (tid < s) red[tid] += red[tid + s];
        __syncthreads();
      }
      if (tid == 0) gacc[d] += red[0] * (1.f / 512.f);
      __syncthreads();
    }
  }

  if (tid < 16) gout[(size_t)(tid * 4 + h) * 512 + i] = gacc[tid];
}

// ---------------------------------------------------------------------------
extern "C" void kernel_launch(void* const* d_in, const int* in_sizes, int n_in,
                              void* d_out, int out_size, void* d_ws,
                              size_t ws_size, hipStream_t stream) {
  (void)in_sizes; (void)n_in; (void)out_size; (void)ws_size;
  const float* X      = (const float*)d_in[0];
  const int*   dist1  = (const int*)d_in[1];
  const int*   dist2  = (const int*)d_in[2];
  const float* W_e1   = (const float*)d_in[3];
  const float* bn_e1  = (const float*)d_in[4];
  const float* W_e1s  = (const float*)d_in[5];
  const float* bn_e1s = (const float*)d_in[6];
  const float* W_h1a  = (const float*)d_in[7];
  const float* bn_h1a = (const float*)d_in[8];
  const float* W_h1b  = (const float*)d_in[9];
  const float* bn_h1b = (const float*)d_in[10];
  const float* W_h1as = (const float*)d_in[11];
  const float* bn_h1as= (const float*)d_in[12];
  const float* W_h1bs = (const float*)d_in[13];
  const float* bn_h1bs= (const float*)d_in[14];
  const float* W_e2   = (const float*)d_in[15];
  const float* bn_e2  = (const float*)d_in[16];
  const float* W_e2s  = (const float*)d_in[17];
  const float* bn_e2s = (const float*)d_in[18];
  const float* w_a1   = (const float*)d_in[19];
  const float* w_a2   = (const float*)d_in[20];
  const float* w_a1s  = (const float*)d_in[21];
  const float* w_a2s  = (const float*)d_in[22];

  float* out = (float*)d_out;
  float* g_out   = out;                      // 64*512
  float* hop_out = out + 64 * 512;           // 2 x 10*512*512

  float* P1 = (float*)d_ws;
  float* Q1 = P1 + 64 * 512;
  float* P2 = Q1 + 64 * 512;
  float* Q2 = P2 + 64 * 512;
  _Float16* e1w1 = (_Float16*)(Q2 + 64 * 512);
  _Float16* e1w2 = e1w1 + (size_t)64 * 512 * 512;

  pq_kernel<<<512, 256, 0, stream>>>(X, W_e1, W_e1s, P1, Q1, P2, Q2);

  e1_attn_kernel<<<dim3(512, 2), 256, 0, stream>>>(
      P1, Q1, P2, Q2, bn_e1, bn_e1s, w_a1, w_a1s, e1w1, e1w2);

  hop_kernel<<<dim3(512, 2), 256, 0, stream>>>(
      e1w1, e1w2, W_h1a, bn_h1a, W_h1b, bn_h1b,
      W_h1as, bn_h1as, W_h1bs, bn_h1bs, hop_out);

  e2_kernel<<<dim3(512, 4), 256, 0, stream>>>(
      e1w1, e1w2, W_e2, bn_e2, w_a2, W_e2s, bn_e2s, w_a2s,
      dist1, dist2, g_out);
}
